// MHFFNMoEDecoderLayer_59219009077893
// MI455X (gfx1250) — matrix operand order repaired and checked by reading, not compiled, not run on hardware
//
#include <hip/hip_runtime.h>
#include <hip/hip_bf16.h>

#define S_LEN 1024
#define HDIM  2048
#define NHN   16
#define NKVH  4
#define HD    128
#define NEXP  8
#define FDIM  256
#define KVW   (NKVH * HD)          // 512
#define EF    (NEXP * FDIM)        // 2048
#define ACT_LD (NHN * EF)          // 32768
#define SCALE_ATTN 0.08838834764831845f
#define BPAD 48                    // padded LDS row stride (halves) for 64x32 B panels

typedef __attribute__((ext_vector_type(16))) _Float16 v16h;
typedef __attribute__((ext_vector_type(8)))  _Float16 v8h;
typedef __attribute__((ext_vector_type(8)))  float    v8f;
typedef __attribute__((vector_size(16)))     int      v4i32;

#ifndef __has_builtin
#define __has_builtin(x) 0
#endif
#if __has_builtin(__builtin_amdgcn_global_load_async_to_lds_b128)
#define HAVE_ASYNC_LDS 1
#else
#define HAVE_ASYNC_LDS 0
#endif

static __device__ __forceinline__ v8f wmma_f16(v16h a, v16h b, v8f c) {
  return __builtin_amdgcn_wmma_f32_16x16x32_f16(false, a, false, b, (short)0, c, false, false);
}

// one 16-byte global->LDS transfer (async on gfx1250 toolchains that expose it)
static __device__ __forceinline__ void stage_b128(const _Float16* g, _Float16* l) {
#if HAVE_ASYNC_LDS
  __builtin_amdgcn_global_load_async_to_lds_b128(
      (__attribute__((address_space(1))) v4i32*)(uintptr_t)g,
      (__attribute__((address_space(3))) v4i32*)(unsigned)(uintptr_t)l, 0, 0);
#else
  *(v8h*)l = *(const v8h*)g;   // global_load_b128 + ds_store_b128 fallback
#endif
}

static __device__ __forceinline__ void wait_async() {
#if HAVE_ASYNC_LDS
#if __has_builtin(__builtin_amdgcn_s_wait_asynccnt)
  __builtin_amdgcn_s_wait_asynccnt(0);
#else
  asm volatile("s_wait_asynccnt 0" ::: "memory");
#endif
#endif
}

// A-fragment: 16x32 f16 tile, rows contiguous-in-K:
// lane m=lane&15, kbase=(lane>>4)*16; halves 0..15 -> K=kb..kb+15
static __device__ __forceinline__ v16h load_a16x32(const _Float16* __restrict__ base, int ld) {
  const int lane = threadIdx.x & 31;
  const _Float16* p = base + (size_t)(lane & 15) * ld + ((lane >> 4) << 4);
  v8h lo = *(const v8h*)p;
  v8h hi = *(const v8h*)(p + 8);
  v16h a;
#pragma unroll
  for (int i = 0; i < 8; ++i) { a[i] = lo[i]; a[8 + i] = hi[i]; }
  return a;
}

// B-fragment: 32x16 (KxN) f16 from a global row-major [N][K] panel
static __device__ __forceinline__ v16h load_b32x16(const _Float16* __restrict__ base, int ld) {
  const int lane = threadIdx.x & 31;
  const _Float16* p = base + (size_t)(lane & 15) * ld + ((lane >> 4) << 4);
  return *(const v16h*)p;
}

// ---------------- conversion kernels ----------------
__global__ void cvt_f16(const float* __restrict__ in, _Float16* __restrict__ out, int n) {
  int i = blockIdx.x * 256 + threadIdx.x;
  if (i < n) out[i] = (_Float16)in[i];
}

// v_w [g][e][f][d] (4,8,256,128) -> vwt [g][d][e*256+f]  (K-contiguous B for FFN GEMM)
__global__ void cvt_vw_t(const float* __restrict__ in, _Float16* __restrict__ out) {
  int i = blockIdx.x * 256 + threadIdx.x;   // 2^20 total
  int d  = i & 127;
  int ef = (i >> 7) & 2047;
  int g  = i >> 18;
  out[((size_t)g * HD + d) * EF + ef] = (_Float16)in[i];
}

// ---------------- norms ----------------
__global__ __launch_bounds__(256) void rmsnorm_f16(const float* __restrict__ x,
                                                   const float* __restrict__ w,
                                                   _Float16* __restrict__ out) {
  __shared__ float red[256];
  const int s = blockIdx.x;
  float ss = 0.f;
  for (int i = threadIdx.x; i < HDIM; i += 256) { float v = x[(size_t)s * HDIM + i]; ss += v * v; }
  red[threadIdx.x] = ss;
  __syncthreads();
  for (int st = 128; st > 0; st >>= 1) {
    if (threadIdx.x < st) red[threadIdx.x] += red[threadIdx.x + st];
    __syncthreads();
  }
  float inv = rsqrtf(red[0] / (float)HDIM + 1e-6f);
  for (int i = threadIdx.x; i < HDIM; i += 256)
    out[(size_t)s * HDIM + i] = (_Float16)(x[(size_t)s * HDIM + i] * inv * w[i]);
}

__global__ __launch_bounds__(256) void rmsnorm_f32(const float* __restrict__ x,
                                                   const float* __restrict__ w,
                                                   float* __restrict__ out) {
  __shared__ float red[256];
  const int s = blockIdx.x;
  float ss = 0.f;
  for (int i = threadIdx.x; i < HDIM; i += 256) { float v = x[(size_t)s * HDIM + i]; ss += v * v; }
  red[threadIdx.x] = ss;
  __syncthreads();
  for (int st = 128; st > 0; st >>= 1) {
    if (threadIdx.x < st) red[threadIdx.x] += red[threadIdx.x + st];
    __syncthreads();
  }
  float inv = rsqrtf(red[0] / (float)HDIM + 1e-6f);
  for (int i = threadIdx.x; i < HDIM; i += 256)
    out[(size_t)s * HDIM + i] = x[(size_t)s * HDIM + i] * inv * w[i];
}

// per-head rmsnorm: one wave per (s, head)
__global__ __launch_bounds__(128) void headnorm(const float* __restrict__ y1,
                                                const float* __restrict__ mhw,
                                                _Float16* __restrict__ y16) {
  const int gw = blockIdx.x * 4 + (threadIdx.x >> 5);
  const int s = gw >> 4, n = gw & 15, lane = threadIdx.x & 31;
  const float* p = y1 + (size_t)s * HDIM + n * HD;
  float ss = 0.f;
#pragma unroll
  for (int i = 0; i < 4; ++i) { float v = p[lane + 32 * i]; ss += v * v; }
#pragma unroll
  for (int off = 16; off; off >>= 1) ss += __shfl_xor(ss, off, 32);
  float inv = rsqrtf(ss / (float)HD + 1e-6f);
#pragma unroll
  for (int i = 0; i < 4; ++i) {
    int d = lane + 32 * i;
    y16[(size_t)s * HDIM + n * HD + d] = (_Float16)(p[d] * inv * mhw[n * HD + d]);
  }
}

// ---------------- RoPE (pairwise, race free) ----------------
__global__ void rope_kernel(_Float16* __restrict__ t, const float* __restrict__ cosb,
                            const float* __restrict__ sinb, int nheads, int total) {
  int idx = blockIdx.x * 256 + threadIdx.x;  // over S*nheads*64
  if (idx >= total) return;
  int d  = idx & 63;
  int sn = idx >> 6;
  int s  = sn / nheads;
  size_t b = (size_t)sn * HD;
  float x1 = (float)t[b + d];
  float x2 = (float)t[b + d + 64];
  float c  = cosb[(size_t)s * HD + d];   // cos[d] == cos[d+64]
  float sv = sinb[(size_t)s * HD + d];
  t[b + d]      = (_Float16)(x1 * c - x2 * sv);
  t[b + d + 64] = (_Float16)(x2 * c + x1 * sv);
}

// =====================================================================
// WMMA GEMM: C[M,N] = A[M,K] * B[N,K]^T with LDS-staged, double-buffered
// B panels (64 rows x 32 K) shared by the 8 waves of the block.
// =====================================================================
__global__ __launch_bounds__(256) void gemm_f16o(const _Float16* __restrict__ A, int lda,
                                                 const _Float16* __restrict__ B, int ldb,
                                                 _Float16* __restrict__ C, int ldc, int K) {
  __shared__ __align__(64) _Float16 bb[2][64 * BPAD];
  const int tid = threadIdx.x;
  const int wave = tid >> 5, lane = tid & 31;
  const int m0 = (blockIdx.y * 8 + wave) * 16;
  const int n0 = blockIdx.x * 64;
  const int row = tid >> 2, seg = tid & 3;          // 64 rows x 4 x 16B segments
  const _Float16* gB = B + (size_t)(n0 + row) * ldb + seg * 8;

  stage_b128(gB, &bb[0][row * BPAD + seg * 8]);
  wait_async();
  __syncthreads();

  v8f zero = {0.f, 0.f, 0.f, 0.f, 0.f, 0.f, 0.f, 0.f};
  v8f acc[4] = {zero, zero, zero, zero};
  for (int k = 0; k < K; k += 32) {
    const int st = (k >> 5) & 1;
    if (k + 32 < K) stage_b128(gB + k + 32, &bb[st ^ 1][row * BPAD + seg * 8]);
    v16h a = load_a16x32(A + (size_t)m0 * lda + k, lda);
    __builtin_prefetch(A + (size_t)m0 * lda + k + 64, 0, 0);
#pragma unroll
    for (int t = 0; t < 4; ++t) {
      const _Float16* p = &bb[st][(t * 16 + (lane & 15)) * BPAD + ((lane >> 4) << 4)];
      v16h b = *(const v16h*)p;
      acc[t] = wmma_f16(a, b, acc[t]);
    }
    wait_async();
    __syncthreads();
  }
  const int colb = lane & 15, rofs = (lane >> 4) << 3;
#pragma unroll
  for (int t = 0; t < 4; ++t)
#pragma unroll
    for (int r = 0; r < 8; ++r)
      C[(size_t)(m0 + rofs + r) * ldc + n0 + t * 16 + colb] = (_Float16)acc[t][r];
}

// output projection + residual (fp32 out), same LDS staging
__global__ __launch_bounds__(256) void gemm_res_f32(const _Float16* __restrict__ A, int lda,
                                                    const _Float16* __restrict__ B, int ldb,
                                                    const float* __restrict__ res,
                                                    float* __restrict__ C, int ldc, int K) {
  __shared__ __align__(64) _Float16 bb[2][64 * BPAD];
  const int tid = threadIdx.x;
  const int wave = tid >> 5, lane = tid & 31;
  const int m0 = (blockIdx.y * 8 + wave) * 16;
  const int n0 = blockIdx.x * 64;
  const int row = tid >> 2, seg = tid & 3;
  const _Float16* gB = B + (size_t)(n0 + row) * ldb + seg * 8;

  stage_b128(gB, &bb[0][row * BPAD + seg * 8]);
  wait_async();
  __syncthreads();

  v8f zero = {0.f, 0.f, 0.f, 0.f, 0.f, 0.f, 0.f, 0.f};
  v8f acc[4] = {zero, zero, zero, zero};
  for (int k = 0; k < K; k += 32) {
    const int st = (k >> 5) & 1;
    if (k + 32 < K) stage_b128(gB + k + 32, &bb[st ^ 1][row * BPAD + seg * 8]);
    v16h a = load_a16x32(A + (size_t)m0 * lda + k, lda);
    __builtin_prefetch(A + (size_t)m0 * lda + k + 64, 0, 0);
#pragma unroll
    for (int t = 0; t < 4; ++t) {
      const _Float16* p = &bb[st][(t * 16 + (lane & 15)) * BPAD + ((lane >> 4) << 4)];
      v16h b = *(const v16h*)p;
      acc[t] = wmma_f16(a, b, acc[t]);
    }
    wait_async();
    __syncthreads();
  }
  const int colb = lane & 15, rofs = (lane >> 4) << 3;
#pragma unroll
  for (int t = 0; t < 4; ++t)
#pragma unroll
    for (int r = 0; r < 8; ++r) {
      size_t idx = (size_t)(m0 + rofs + r) * ldc + n0 + t * 16 + colb;
      C[idx] = res[idx] + acc[t][r];
    }
}

// ---------------- flash attention: one wave per (head, 16-query tile) ----------------
__global__ __launch_bounds__(128) void attn_fa(const _Float16* __restrict__ q16,
                                               const _Float16* __restrict__ k16,
                                               const _Float16* __restrict__ v16,
                                               _Float16* __restrict__ att16) {
  __shared__ __align__(32) _Float16 pbuf[4][16 * 32];
  const int wave = threadIdx.x >> 5;
  const int lane = threadIdx.x & 31;
  const int gw = blockIdx.x * 4 + wave;
  const int h  = gw & (NHN - 1);
  const int q0 = (gw >> 4) * 16;
  const int g  = h >> 2;
  const _Float16* Q = q16 + (size_t)q0 * HDIM + h * HD;
  const _Float16* K = k16 + g * HD;
  const _Float16* V = v16 + g * HD;

  v16h qa[4];
#pragma unroll
  for (int c = 0; c < 4; ++c) qa[c] = load_a16x32(Q + c * 32, HDIM);

  v8f zero = {0.f, 0.f, 0.f, 0.f, 0.f, 0.f, 0.f, 0.f};
  v8f o[8];
  float m_run[8], l_run[8];
#pragma unroll
  for (int r = 0; r < 8; ++r) { o[r] = zero; m_run[r] = -3.0e38f; l_run[r] = 0.f; }

  const int colb = lane & 15;
  const int rofs = (lane >> 4) << 3;

  for (int kb = 0; kb < q0 + 16; kb += 32) {
    v8f s0 = zero, s1 = zero;
#pragma unroll
    for (int c = 0; c < 4; ++c) {
      v16h b0 = load_b32x16(K + (size_t)kb * KVW + c * 32, KVW);
      v16h b1 = load_b32x16(K + (size_t)(kb + 16) * KVW + c * 32, KVW);
      s0 = wmma_f16(qa[c], b0, s0);
      s1 = wmma_f16(qa[c], b1, s1);
    }
#pragma unroll
    for (int r = 0; r < 8; ++r) {
      int qrow = q0 + rofs + r;
      float a0 = (kb + colb      <= qrow) ? s0[r] * SCALE_ATTN : -3.0e38f;
      float a1 = (kb + 16 + colb <= qrow) ? s1[r] * SCALE_ATTN : -3.0e38f;
      float mx = fmaxf(a0, a1);
#pragma unroll
      for (int off = 8; off; off >>= 1) mx = fmaxf(mx, __shfl_xor(mx, off, 32));
      float mnew = fmaxf(m_run[r], mx);
      float corr = __expf(m_run[r] - mnew);
      float p0 = __expf(a0 - mnew);
      float p1 = __expf(a1 - mnew);
      float ls = p0 + p1;
#pragma unroll
      for (int off = 8; off; off >>= 1) ls += __shfl_xor(ls, off, 32);
      l_run[r] = l_run[r] * corr + ls;
      m_run[r] = mnew;
#pragma unroll
      for (int t = 0; t < 8; ++t) o[t][r] *= corr;
      pbuf[wave][(rofs + r) * 32 + colb]      = (_Float16)p0;
      pbuf[wave][(rofs + r) * 32 + colb + 16] = (_Float16)p1;
    }
    // reload P (16x32) as A-fragment from LDS (same-wave DS ops are in order)
    const _Float16* ap = &pbuf[wave][(lane & 15) * 32 + ((lane >> 4) << 4)];
    v16h pa;
#pragma unroll
    for (int i = 0; i < 8; ++i) { pa[i] = ap[i]; pa[8 + i] = ap[8 + i]; }
#pragma unroll
    for (int t = 0; t < 8; ++t) {
      const _Float16* vp = V + (size_t)(kb + ((lane >> 4) << 4)) * KVW + t * 16 + (lane & 15);
      v16h vb;
#pragma unroll
      for (int i = 0; i < 16; ++i) vb[i] = vp[(size_t)i * KVW];
      o[t] = wmma_f16(pa, vb, o[t]);
    }
  }
#pragma unroll
  for (int r = 0; r < 8; ++r) {
    float inv = 1.0f / l_run[r];
#pragma unroll
    for (int t = 0; t < 8; ++t)
      att16[(size_t)(q0 + rofs + r) * HDIM + h * HD + t * 16 + colb] = (_Float16)(o[t][r] * inv);
  }
}

// ---------------- router: one wave per (s, head) ----------------
__global__ __launch_bounds__(128) void router_topk(const _Float16* __restrict__ y16,
                                                   const float* __restrict__ rw,
                                                   float* __restrict__ wdense) {
  const int gw = blockIdx.x * 4 + (threadIdx.x >> 5);
  const int s = gw >> 4, n = gw & 15, lane = threadIdx.x & 31;
  const _Float16* yp = y16 + (size_t)s * HDIM + n * HD;
  float lg[NEXP];
#pragma unroll
  for (int e = 0; e < NEXP; ++e) {
    const float* wp = rw + (size_t)(n * NEXP + e) * HD;
    float d = 0.f;
#pragma unroll
    for (int i = 0; i < 4; ++i) d += (float)yp[lane + 32 * i] * wp[lane + 32 * i];
#pragma unroll
    for (int off = 16; off; off >>= 1) d += __shfl_xor(d, off, 32);
    lg[e] = d;
  }
  float mx = lg[0];
#pragma unroll
  for (int e = 1; e < NEXP; ++e) mx = fmaxf(mx, lg[e]);
  float ex[NEXP];
#pragma unroll
  for (int e = 0; e < NEXP; ++e) ex[e] = __expf(lg[e] - mx);
  int i1 = 0; float v1 = ex[0];
#pragma unroll
  for (int e = 1; e < NEXP; ++e) if (ex[e] > v1) { v1 = ex[e]; i1 = e; }
  int i2 = (i1 == 0) ? 1 : 0; float v2 = ex[i2];
#pragma unroll
  for (int e = 0; e < NEXP; ++e)
    if (e != i1 && ex[e] > v2) { v2 = ex[e]; i2 = e; }
  float tot = v1 + v2;
  if (lane < NEXP) {
    float w = (lane == i1) ? v1 / tot : ((lane == i2) ? v2 / tot : 0.f);
    wdense[((size_t)s * NHN + n) * NEXP + lane] = w;
  }
}

// ---------------- gate/up GEMM + SiLU*up*gate_weight epilogue (dual LDS panels) ----------------
__global__ __launch_bounds__(256) void gemm_gateup(const _Float16* __restrict__ y16,
                                                   const _Float16* __restrict__ kw16,
                                                   const _Float16* __restrict__ uw16,
                                                   const float* __restrict__ wdense,
                                                   _Float16* __restrict__ act16) {
  __shared__ __align__(64) _Float16 bg[2][64 * BPAD];
  __shared__ __align__(64) _Float16 bu[2][64 * BPAD];
  const int tid = threadIdx.x;
  const int wave = tid >> 5, lane = tid & 31;
  const int ne = blockIdx.z;
  const int n = ne >> 3, e = ne & 7, g = n >> 2;
  const _Float16* A  = y16 + n * HD;                                 // lda = HDIM
  const int m0 = (blockIdx.y * 8 + wave) * 16;
  const int f0 = blockIdx.x * 64;
  const int row = tid >> 2, seg = tid & 3;
  const _Float16* gBg = kw16 + (size_t)(g * NEXP + e) * FDIM * HD + (size_t)(f0 + row) * HD + seg * 8;
  const _Float16* gBu = uw16 + (size_t)(g * NEXP + e) * FDIM * HD + (size_t)(f0 + row) * HD + seg * 8;

  stage_b128(gBg, &bg[0][row * BPAD + seg * 8]);
  stage_b128(gBu, &bu[0][row * BPAD + seg * 8]);
  wait_async();
  __syncthreads();

  v8f zero = {0.f, 0.f, 0.f, 0.f, 0.f, 0.f, 0.f, 0.f};
  v8f ag[4] = {zero, zero, zero, zero};
  v8f au[4] = {zero, zero, zero, zero};
  for (int k = 0; k < HD; k += 32) {
    const int st = (k >> 5) & 1;
    if (k + 32 < HD) {
      stage_b128(gBg + k + 32, &bg[st ^ 1][row * BPAD + seg * 8]);
      stage_b128(gBu + k + 32, &bu[st ^ 1][row * BPAD + seg * 8]);
    }
    v16h a = load_a16x32(A + (size_t)m0 * HDIM + k, HDIM);
#pragma unroll
    for (int t = 0; t < 4; ++t) {
      const _Float16* pg = &bg[st][(t * 16 + (lane & 15)) * BPAD + ((lane >> 4) << 4)];
      const _Float16* pu = &bu[st][(t * 16 + (lane & 15)) * BPAD + ((lane >> 4) << 4)];
      ag[t] = wmma_f16(a, *(const v16h*)pg, ag[t]);
      au[t] = wmma_f16(a, *(const v16h*)pu, au[t]);
    }
    wait_async();
    __syncthreads();
  }
  const int colb = lane & 15, rofs = (lane >> 4) << 3;
#pragma unroll
  for (int t = 0; t < 4; ++t)
#pragma unroll
    for (int r = 0; r < 8; ++r) {
      int s = m0 + rofs + r;
      float w = wdense[((size_t)s * NHN + n) * NEXP + e];
      float gv = ag[t][r];
      float uv = au[t][r];
      float av = gv / (1.f + __expf(-gv)) * uv * w;
      act16[(size_t)s * ACT_LD + n * EF + e * FDIM + f0 + t * 16 + colb] = (_Float16)av;
    }
}

// ---------------- FFN down GEMM + final residual -> d_out (LDS-staged B) ----------------
__global__ __launch_bounds__(256) void gemm_ffn_out(const _Float16* __restrict__ act16,
                                                    const _Float16* __restrict__ vwt16,
                                                    const float* __restrict__ h32,
                                                    float* __restrict__ out) {
  __shared__ __align__(64) _Float16 bb[2][64 * BPAD];
  const int tid = threadIdx.x;
  const int wave = tid >> 5, lane = tid & 31;
  const int n = blockIdx.z, g = n >> 2;
  const _Float16* A = act16 + n * EF;                       // lda = ACT_LD
  const int m0 = (blockIdx.y * 8 + wave) * 16;
  const int d0 = blockIdx.x * 64;
  const int row = tid >> 2, seg = tid & 3;
  const _Float16* gB = vwt16 + (size_t)g * HD * EF + (size_t)(d0 + row) * EF + seg * 8;

  stage_b128(gB, &bb[0][row * BPAD + seg * 8]);
  wait_async();
  __syncthreads();

  v8f zero = {0.f, 0.f, 0.f, 0.f, 0.f, 0.f, 0.f, 0.f};
  v8f acc[4] = {zero, zero, zero, zero};
  for (int k = 0; k < EF; k += 32) {
    const int st = (k >> 5) & 1;
    if (k + 32 < EF) stage_b128(gB + k + 32, &bb[st ^ 1][row * BPAD + seg * 8]);
    v16h a = load_a16x32(A + (size_t)m0 * ACT_LD + k, ACT_LD);
    __builtin_prefetch(A + (size_t)m0 * ACT_LD + k + 64, 0, 0);
#pragma unroll
    for (int t = 0; t < 4; ++t) {
      const _Float16* p = &bb[st][(t * 16 + (lane & 15)) * BPAD + ((lane >> 4) << 4)];
      v16h b = *(const v16h*)p;
      acc[t] = wmma_f16(a, b, acc[t]);
    }
    wait_async();
    __syncthreads();
  }
  const int colb = lane & 15, rofs = (lane >> 4) << 3;
#pragma unroll
  for (int t = 0; t < 4; ++t)
#pragma unroll
    for (int r = 0; r < 8; ++r) {
      size_t idx = (size_t)(m0 + rofs + r) * HDIM + n * HD + d0 + t * 16 + colb;
      out[idx] = h32[idx] + acc[t][r];
    }
}

// ---------------- launch ----------------
extern "C" void kernel_launch(void* const* d_in, const int* in_sizes, int n_in,
                              void* d_out, int out_size, void* d_ws, size_t ws_size,
                              hipStream_t stream) {
  const float* hidden = (const float*)d_in[0];
  const float* cosb   = (const float*)d_in[1];
  const float* sinb   = (const float*)d_in[2];
  /* d_in[3] attn_mask: causal applied analytically */
  const float* ln1w   = (const float*)d_in[4];
  const float* ln2w   = (const float*)d_in[5];
  const float* wq     = (const float*)d_in[6];
  const float* wk     = (const float*)d_in[7];
  const float* wv     = (const float*)d_in[8];
  const float* wo     = (const float*)d_in[9];
  const float* mhw    = (const float*)d_in[10];
  const float* rw     = (const float*)d_in[11];
  const float* kw     = (const float*)d_in[12];
  const float* uw     = (const float*)d_in[13];
  const float* vw     = (const float*)d_in[14];
  float* out = (float*)d_out;

  char* ws = (char*)d_ws;
  size_t off = 0;
  auto carve = [&](size_t bytes) { char* p = ws + off; off += (bytes + 255) & ~(size_t)255; return p; };
  _Float16* X16   = (_Float16*)carve((size_t)S_LEN * HDIM * 2);
  _Float16* WQ16  = (_Float16*)carve((size_t)HDIM * HDIM * 2);
  _Float16* WK16  = (_Float16*)carve((size_t)KVW * HDIM * 2);
  _Float16* WV16  = (_Float16*)carve((size_t)KVW * HDIM * 2);
  _Float16* WO16  = (_Float16*)carve((size_t)HDIM * HDIM * 2);
  _Float16* Q16   = (_Float16*)carve((size_t)S_LEN * HDIM * 2);
  _Float16* K16   = (_Float16*)carve((size_t)S_LEN * KVW * 2);
  _Float16* V16   = (_Float16*)carve((size_t)S_LEN * KVW * 2);
  _Float16* ATT16 = (_Float16*)carve((size_t)S_LEN * HDIM * 2);
  float*    H32   = (float*)carve((size_t)S_LEN * HDIM * 4);
  float*    Y132  = (float*)carve((size_t)S_LEN * HDIM * 4);
  _Float16* Y16   = (_Float16*)carve((size_t)S_LEN * HDIM * 2);
  float*    WDEN  = (float*)carve((size_t)S_LEN * NHN * NEXP * 4);
  _Float16* KW16  = (_Float16*)carve((size_t)NKVH * NEXP * FDIM * HD * 2);
  _Float16* UW16  = (_Float16*)carve((size_t)NKVH * NEXP * FDIM * HD * 2);
  _Float16* VWT16 = (_Float16*)carve((size_t)NKVH * HD * EF * 2);
  _Float16* ACT16 = (_Float16*)carve((size_t)S_LEN * ACT_LD * 2);

  // 1) weight conversions
  cvt_f16<<<(HDIM * HDIM) / 256, 256, 0, stream>>>(wq, WQ16, HDIM * HDIM);
  cvt_f16<<<(KVW * HDIM) / 256, 256, 0, stream>>>(wk, WK16, KVW * HDIM);
  cvt_f16<<<(KVW * HDIM) / 256, 256, 0, stream>>>(wv, WV16, KVW * HDIM);
  cvt_f16<<<(HDIM * HDIM) / 256, 256, 0, stream>>>(wo, WO16, HDIM * HDIM);
  cvt_f16<<<(NKVH * NEXP * FDIM * HD) / 256, 256, 0, stream>>>(kw, KW16, NKVH * NEXP * FDIM * HD);
  cvt_f16<<<(NKVH * NEXP * FDIM * HD) / 256, 256, 0, stream>>>(uw, UW16, NKVH * NEXP * FDIM * HD);
  cvt_vw_t<<<(NKVH * NEXP * FDIM * HD) / 256, 256, 0, stream>>>(vw, VWT16);

  // 2) rmsnorm1 -> f16
  rmsnorm_f16<<<S_LEN, 256, 0, stream>>>(hidden, ln1w, X16);

  // 3) QKV projections (WMMA, async-LDS staged B)
  gemm_f16o<<<dim3(HDIM / 64, S_LEN / 128), 256, 0, stream>>>(X16, HDIM, WQ16, HDIM, Q16, HDIM, HDIM);
  gemm_f16o<<<dim3(KVW / 64, S_LEN / 128), 256, 0, stream>>>(X16, HDIM, WK16, HDIM, K16, KVW, HDIM);
  gemm_f16o<<<dim3(KVW / 64, S_LEN / 128), 256, 0, stream>>>(X16, HDIM, WV16, HDIM, V16, KVW, HDIM);

  // 4) RoPE on q and k
  rope_kernel<<<(S_LEN * NHN * 64) / 256, 256, 0, stream>>>(Q16, cosb, sinb, NHN, S_LEN * NHN * 64);
  rope_kernel<<<(S_LEN * NKVH * 64) / 256, 256, 0, stream>>>(K16, cosb, sinb, NKVH, S_LEN * NKVH * 64);

  // 5) flash attention (WMMA QK^T and PV)
  attn_fa<<<(S_LEN / 16) * NHN / 4, 128, 0, stream>>>(Q16, K16, V16, ATT16);

  // 6) output projection + residual -> h (fp32)
  gemm_res_f32<<<dim3(HDIM / 64, S_LEN / 128), 256, 0, stream>>>(ATT16, HDIM, WO16, HDIM, hidden, H32, HDIM, HDIM);

  // 7) rmsnorm2 then per-head rmsnorm -> y (f16)
  rmsnorm_f32<<<S_LEN, 256, 0, stream>>>(H32, ln2w, Y132);
  headnorm<<<(S_LEN * NHN) / 4, 128, 0, stream>>>(Y132, mhw, Y16);

  // 8) router softmax + top-2 -> dense gate weights
  router_topk<<<(S_LEN * NHN) / 4, 128, 0, stream>>>(Y16, rw, WDEN);

  // 9) gate/up GEMMs with fused SiLU epilogue -> act (f16)
  gemm_gateup<<<dim3(FDIM / 64, S_LEN / 128, NHN * NEXP), 256, 0, stream>>>(Y16, KW16, UW16, WDEN, ACT16);

  // 10) FFN down GEMM + final residual -> d_out (fp32)
  gemm_ffn_out<<<dim3(HD / 64, S_LEN / 128, NHN), 256, 0, stream>>>(ACT16, VWT16, H32, out);

  (void)in_sizes; (void)n_in; (void)out_size; (void)ws_size;
}
